// SimplifiedCachedDecoderWrapper_66709432041893
// MI455X (gfx1250) — matrix operand order repaired and checked by reading, not compiled, not run on hardware
//
#include <hip/hip_runtime.h>
#include <hip/hip_bf16.h>

// ---------------- model dims ----------------
#define L_     8
#define NH_    16
#define DH_    64
#define H_     1024
#define FFN_   4096
#define ENC_   1500
#define ENCP_  1504     // padded to multiple of 32 for WMMA K loop
#define VOCAB_ 8192
#define SMAX_  2048

typedef __attribute__((ext_vector_type(16))) __bf16          v16bf;
typedef __attribute__((ext_vector_type(8)))  float           v8f;
typedef __attribute__((ext_vector_type(8)))  unsigned short  us8;

__device__ __forceinline__ unsigned short f2bf(float f) {
    unsigned u = __float_as_uint(f);
    unsigned r = (u + 0x7FFFu + ((u >> 16) & 1u)) >> 16;   // RNE
    return (unsigned short)r;
}

// Build a v16bf WMMA fragment from two 16-byte (8 x bf16) vector loads.
__device__ __forceinline__ v16bf ldfrag(const unsigned short* p0, const unsigned short* p1) {
    union { us8 h[2]; v16bf v; } u;
    u.h[0] = *(const us8*)p0;
    u.h[1] = *(const us8*)p1;
    return u.v;
}

// ---------------- embed: x = tok_emb[id] + pos_emb[step] ----------------
__global__ __launch_bounds__(256) void k_embed(const float* tok, const float* pos,
                                               const int* id, const int* stepp, float* x) {
    int i = blockIdx.x * 256 + threadIdx.x;
    if (i < H_) x[i] = tok[(size_t)id[0] * H_ + i] + pos[(size_t)stepp[0] * H_ + i];
}

// ---------------- one-time: enc -> bf16 row-major (padded) + bf16 transposed ----------------
// enc_bf : ENCP_ x H_   (rows >= ENC_ zero)   -- A operand of scores GEMM
// enc_t  : H_ x ENCP_   (cols >= ENC_ zero)   -- B operand of wsum GEMM
__global__ __launch_bounds__(256) void k_prep_enc(const float* __restrict__ enc,
                                                  unsigned short* __restrict__ enc_bf,
                                                  unsigned short* __restrict__ enc_t) {
    __shared__ unsigned short tile[32][33];
    int tx = threadIdx.x & 31, ty = threadIdx.x >> 5;    // 32 x 8
    int f0 = blockIdx.y * 32, j0 = blockIdx.x * 32;
#pragma unroll
    for (int r = 0; r < 4; ++r) {
        int f = f0 + ty + r * 8, j = j0 + tx;
        unsigned short b = (f < ENC_) ? f2bf(enc[(size_t)f * H_ + j]) : (unsigned short)0;
        enc_bf[(size_t)f * H_ + j] = b;
        tile[ty + r * 8][tx] = b;
    }
    __syncthreads();
#pragma unroll
    for (int r = 0; r < 4; ++r) {
        int j = j0 + ty + r * 8, f = f0 + tx;
        enc_t[(size_t)j * ENCP_ + f] = tile[tx][ty + r * 8];
    }
}

// ---------------- layernorm over H=1024, one block ----------------
__global__ __launch_bounds__(256) void k_layernorm(const float* x, const float* g,
                                                   const float* b, float* out) {
    __shared__ float red[256];
    int tid = threadIdx.x;
    float s = 0.f;
    for (int i = tid; i < H_; i += 256) s += x[i];
    red[tid] = s; __syncthreads();
    for (int o = 128; o > 0; o >>= 1) { if (tid < o) red[tid] += red[tid + o]; __syncthreads(); }
    float m = red[0] / (float)H_; __syncthreads();
    float v = 0.f;
    for (int i = tid; i < H_; i += 256) { float d = x[i] - m; v += d * d; }
    red[tid] = v; __syncthreads();
    for (int o = 128; o > 0; o >>= 1) { if (tid < o) red[tid] += red[tid + o]; __syncthreads(); }
    float r = rsqrtf(red[0] / (float)H_ + 1e-5f); __syncthreads();
    for (int i = tid; i < H_; i += 256) out[i] = (x[i] - m) * r * g[i] + b[i];
}

// ---------------- split-K GEMV: part[s*N+j] = sum_{i in slice s} x[i]*W[i*N+j] ----------------
__global__ __launch_bounds__(256) void k_gemv_part(const float* __restrict__ x,
                                                   const float* __restrict__ W,
                                                   float* __restrict__ part,
                                                   int K, int N, int S) {
    __shared__ float xs[1024];
    int s = blockIdx.y, tid = threadIdx.x;
    int slice = K / S, i0 = s * slice;
    for (int i = tid; i < slice; i += 256) xs[i] = x[i0 + i];
    __syncthreads();
    int j = (blockIdx.x * 256 + tid) * 4;
    const float* Wp = W + (size_t)i0 * N + j;
    float4 acc = make_float4(0.f, 0.f, 0.f, 0.f);
    for (int i = 0; i < slice; ++i) {
        float4 w = *(const float4*)(Wp + (size_t)i * N);
        float xv = xs[i];
        acc.x += xv * w.x; acc.y += xv * w.y; acc.z += xv * w.z; acc.w += xv * w.w;
    }
    *(float4*)&part[(size_t)s * N + j] = acc;
}

// mode: 0=store, 1=accumulate into out, 2=relu(store)
__global__ __launch_bounds__(256) void k_gemv_reduce(const float* part, float* out,
                                                     int N, int S, int mode) {
    int j = blockIdx.x * 256 + threadIdx.x;
    if (j >= N) return;
    float a = 0.f;
    for (int s = 0; s < S; ++s) a += part[(size_t)s * N + j];
    if (mode == 1) out[j] += a;
    else if (mode == 2) out[j] = fmaxf(a, 0.f);
    else out[j] = a;
}

// ---------------- self attention: one block per head ----------------
__global__ __launch_bounds__(256) void k_self_attn(const float* q, const float* Kc,
                                                   const float* Vc, const float* knew,
                                                   const float* vnew, const int* stepp,
                                                   float* out) {
    __shared__ float sq[DH_];
    __shared__ float p[SMAX_];
    __shared__ float red[256];
    __shared__ float acc4[4 * DH_];
    int h = blockIdx.x, tid = threadIdx.x;
    int step = stepp[0], T = step + 1;
    if (tid < DH_) sq[tid] = q[h * DH_ + tid];
    __syncthreads();
    float lmax = -1e30f;
    for (int t = tid; t < T; t += 256) {
        const float* kr = (t < step) ? (Kc + ((size_t)h * SMAX_ + t) * DH_) : (knew + h * DH_);
        float s = 0.f;
        for (int d = 0; d < DH_; d += 4) {
            float4 kv = *(const float4*)(kr + d);
            s += sq[d] * kv.x + sq[d + 1] * kv.y + sq[d + 2] * kv.z + sq[d + 3] * kv.w;
        }
        s *= 0.125f;                 // 1/sqrt(64)
        p[t] = s; lmax = fmaxf(lmax, s);
    }
    red[tid] = lmax; __syncthreads();
    for (int o = 128; o > 0; o >>= 1) { if (tid < o) red[tid] = fmaxf(red[tid], red[tid + o]); __syncthreads(); }
    float M = red[0]; __syncthreads();
    float lsum = 0.f;
    for (int t = tid; t < T; t += 256) { float e = __expf(p[t] - M); p[t] = e; lsum += e; }
    red[tid] = lsum; __syncthreads();
    for (int o = 128; o > 0; o >>= 1) { if (tid < o) red[tid] += red[tid + o]; __syncthreads(); }
    float inv = 1.f / red[0]; __syncthreads();
    int d = tid & 63, grp = tid >> 6;
    float a = 0.f;
    for (int t = grp; t < T; t += 4) {
        const float* vr = (t < step) ? (Vc + ((size_t)h * SMAX_ + t) * DH_) : (vnew + h * DH_);
        a += p[t] * vr[d];
    }
    acc4[grp * DH_ + d] = a; __syncthreads();
    if (tid < DH_)
        out[h * DH_ + tid] = (acc4[tid] + acc4[DH_ + tid] + acc4[2 * DH_ + tid] + acc4[3 * DH_ + tid]) * inv;
}

// ---------------- Ut[h, i] = Ck[i, h*64:(h+1)*64] . qc[h*64:(h+1)*64]  (bf16, K-contiguous) ----------------
__global__ __launch_bounds__(256) void k_crossU(const float* qc, const float* Ck,
                                                unsigned short* Ut) {
    __shared__ float sqc[H_];
    int tid = threadIdx.x;
    for (int i = tid; i < H_; i += 256) sqc[i] = qc[i];
    __syncthreads();
    int gid = blockIdx.x * 256 + tid;          // 0..16383
    int i = gid >> 4, h = gid & 15;
    const float* row = Ck + (size_t)i * H_ + h * DH_;
    const float* qv  = sqc + h * DH_;
    float s = 0.f;
    for (int d = 0; d < DH_; d += 4) {
        float4 w = *(const float4*)(row + d);
        s += qv[d] * w.x + qv[d + 1] * w.y + qv[d + 2] * w.z + qv[d + 3] * w.w;
    }
    Ut[(size_t)h * H_ + i] = f2bf(s);
}

// ---------------- S_t(16 x ENCP) = (enc(ENC x H) @ U(H x 16))^T ----------------
// one wave per 16-row tile of enc; fragments = 2x b128 loads per operand; no conversions
__global__ __launch_bounds__(32) void k_scores_wmma(const unsigned short* __restrict__ enc_bf,
                                                    const unsigned short* __restrict__ Ut,
                                                    float* __restrict__ S_t) {
    int lane = threadIdx.x;
    int g = lane >> 4, ln = lane & 15;
    int m0 = blockIdx.x * 16;
    const unsigned short* arow = enc_bf + (size_t)(m0 + ln) * H_;   // A row (enc frame)
    const unsigned short* brow = Ut + (size_t)ln * H_;              // B col (head), K-contiguous
    v8f c = {};
    for (int k0 = 0; k0 < H_; k0 += 32) {
        v16bf a = ldfrag(arow + k0 + g * 8, arow + k0 + 16 + g * 8);
        v16bf b = ldfrag(brow + k0 + g * 8, brow + k0 + 16 + g * 8);
        c = __builtin_amdgcn_wmma_f32_16x16x32_bf16(false, a, false, b, (short)0, c, false, false);
    }
    // D: lane column = head ln, rows m0+8g .. m0+8g+7 contiguous -> two float4 stores
    float* o = S_t + (size_t)ln * ENCP_ + m0 + 8 * g;
    *(float4*)o       = make_float4(c[0], c[1], c[2], c[3]);
    *(float4*)(o + 4) = make_float4(c[4], c[5], c[6], c[7]);
}

// ---------------- per-head softmax over ENC (scale+mask applied here), emit bf16 Pt[h][ENCP] ----------------
__global__ __launch_bounds__(256) void k_softmax_head(const float* S_t, const float* cmask,
                                                      unsigned short* Pt) {
    __shared__ float red[256];
    int h = blockIdx.x, tid = threadIdx.x;
    const float* row = S_t + (size_t)h * ENCP_;
    float lmax = -1e30f;
    for (int f = tid; f < ENC_; f += 256) lmax = fmaxf(lmax, row[f] * 0.125f + cmask[f]);
    red[tid] = lmax; __syncthreads();
    for (int o = 128; o > 0; o >>= 1) { if (tid < o) red[tid] = fmaxf(red[tid], red[tid + o]); __syncthreads(); }
    float M = red[0]; __syncthreads();
    float lsum = 0.f;
    for (int f = tid; f < ENC_; f += 256) lsum += __expf(row[f] * 0.125f + cmask[f] - M);
    red[tid] = lsum; __syncthreads();
    for (int o = 128; o > 0; o >>= 1) { if (tid < o) red[tid] += red[tid + o]; __syncthreads(); }
    float inv = 1.f / red[0]; __syncthreads();
    for (int f = tid; f < ENCP_; f += 256)
        Pt[(size_t)h * ENCP_ + f] =
            (f < ENC_) ? f2bf(__expf(row[f] * 0.125f + cmask[f] - M) * inv) : (unsigned short)0;
}

// ---------------- Wsum_t(H x 16) = (P^T(16 x ENCP) @ enc(ENCP x H))^T ----------------
__global__ __launch_bounds__(32) void k_wsum_wmma(const unsigned short* __restrict__ Pt,
                                                  const unsigned short* __restrict__ enc_t,
                                                  float* __restrict__ Wsum_t) {
    int lane = threadIdx.x;
    int g = lane >> 4, ln = lane & 15;
    int n0 = blockIdx.x * 16;
    const unsigned short* arow = Pt + (size_t)ln * ENCP_;            // A row = head, K-contiguous
    const unsigned short* brow = enc_t + (size_t)(n0 + ln) * ENCP_;  // B col = H column, K-contiguous
    v8f c = {};
    for (int k0 = 0; k0 < ENCP_; k0 += 32) {
        v16bf a = ldfrag(arow + k0 + g * 8, arow + k0 + 16 + g * 8);
        v16bf b = ldfrag(brow + k0 + g * 8, brow + k0 + 16 + g * 8);
        c = __builtin_amdgcn_wmma_f32_16x16x32_bf16(false, a, false, b, (short)0, c, false, false);
    }
    // D: lane column = H index n0+ln, rows (heads) 8g..8g+7 contiguous in Wsum_t[col][head]
    float* o = Wsum_t + (size_t)(n0 + ln) * NH_ + 8 * g;
    *(float4*)o       = make_float4(c[0], c[1], c[2], c[3]);
    *(float4*)(o + 4) = make_float4(c[4], c[5], c[6], c[7]);
}

// ---------------- out[j] = sum_i Wsum_t[i][j>>6] * Cv[i*H + j] ----------------
__global__ __launch_bounds__(256) void k_crossV(const float* __restrict__ Wsum_t,
                                                const float* __restrict__ Cv,
                                                float* __restrict__ out) {
    int j = blockIdx.x * 256 + threadIdx.x;     // 0..1023
    int h = j >> 6;
    float s = 0.f;
    for (int i = 0; i < H_; ++i) s += Wsum_t[i * NH_ + h] * Cv[(size_t)i * H_ + j];
    out[j] = s;
}

// ---------------- log_softmax over VOCAB ----------------
__global__ __launch_bounds__(256) void k_logsoftmax(const float* lg, float* out) {
    __shared__ float red[256];
    int tid = threadIdx.x;
    float lmax = -1e30f;
    for (int j = tid; j < VOCAB_; j += 256) lmax = fmaxf(lmax, lg[j]);
    red[tid] = lmax; __syncthreads();
    for (int o = 128; o > 0; o >>= 1) { if (tid < o) red[tid] = fmaxf(red[tid], red[tid + o]); __syncthreads(); }
    float M = red[0]; __syncthreads();
    float lsum = 0.f;
    for (int j = tid; j < VOCAB_; j += 256) lsum += __expf(lg[j] - M);
    red[tid] = lsum; __syncthreads();
    for (int o = 128; o > 0; o >>= 1) { if (tid < o) red[tid] += red[tid + o]; __syncthreads(); }
    float lz = __logf(red[0]); __syncthreads();
    for (int j = tid; j < VOCAB_; j += 256) out[j] = lg[j] - M - lz;
}

// ---------------- cache restack: old[:step] ++ new ++ zeros, float4 wide ----------------
__global__ __launch_bounds__(256) void k_cache(const float* __restrict__ ck,
                                               const float* __restrict__ cv,
                                               const float* __restrict__ kn,
                                               const float* __restrict__ vn,
                                               const int* stepp,
                                               float* __restrict__ outk,
                                               float* __restrict__ outv) {
    const long long total = (long long)L_ * NH_ * SMAX_ * DH_ / 4;   // float4 count
    int step = stepp[0];
    for (long long i4 = (long long)blockIdx.x * 256 + threadIdx.x; i4 < total;
         i4 += (long long)gridDim.x * 256) {
        int v  = (int)(i4 & 15);             // float4 within 64-wide d
        int t  = (int)((i4 >> 4) & (SMAX_ - 1));
        int lh = (int)(i4 >> 15);            // (l*16+h)
        float4 kval, vval;
        if (t < step)       { kval = ((const float4*)ck)[i4]; vval = ((const float4*)cv)[i4]; }
        else if (t == step) { kval = ((const float4*)kn)[lh * 16 + v]; vval = ((const float4*)vn)[lh * 16 + v]; }
        else                { kval = make_float4(0.f, 0.f, 0.f, 0.f); vval = kval; }
        ((float4*)outk)[i4] = kval;
        ((float4*)outv)[i4] = vval;
    }
}

// ---------------- host orchestration ----------------
static inline void gemv(hipStream_t st, const float* x, const float* W, float* part,
                        float* out, int K, int N, int S, int mode) {
    k_gemv_part<<<dim3(N / 1024, S), 256, 0, st>>>(x, W, part, K, N, S);
    k_gemv_reduce<<<N / 256, 256, 0, st>>>(part, out, N, S, mode);
}

extern "C" void kernel_launch(void* const* d_in, const int* in_sizes, int n_in,
                              void* d_out, int out_size, void* d_ws, size_t ws_size,
                              hipStream_t stream) {
    const int*   input_id = (const int*)d_in[0];
    const float* enc      = (const float*)d_in[1];
    const float* cache_k  = (const float*)d_in[2];
    const float* cache_v  = (const float*)d_in[3];
    const int*   stepp    = (const int*)d_in[4];
    const float* cmask    = (const float*)d_in[5];
    const float* tok_emb  = (const float*)d_in[6];
    const float* pos_emb  = (const float*)d_in[7];
    const float* Wq = (const float*)d_in[8],  *Wk = (const float*)d_in[9];
    const float* Wv = (const float*)d_in[10], *Wo = (const float*)d_in[11];
    const float* Cq = (const float*)d_in[12], *Ck = (const float*)d_in[13];
    const float* Cv = (const float*)d_in[14], *Co = (const float*)d_in[15];
    const float* W1 = (const float*)d_in[16], *W2 = (const float*)d_in[17];
    const float* ln1_g = (const float*)d_in[18], *ln1_b = (const float*)d_in[19];
    const float* ln2_g = (const float*)d_in[20], *ln2_b = (const float*)d_in[21];
    const float* ln3_g = (const float*)d_in[22], *ln3_b = (const float*)d_in[23];
    const float* lnf_g = (const float*)d_in[24], *lnf_b = (const float*)d_in[25];
    const float* Wout  = (const float*)d_in[26];
    (void)in_sizes; (void)n_in; (void)out_size; (void)ws_size;

    float* w = (float*)d_ws;
    float* x      = w;                 // 1024
    float* h      = w + 1024;          // 1024
    float* q      = w + 2048;          // 1024
    float* t1024  = w + 3072;          // 1024 (attn out / cross tmp)
    float* qc     = w + 4096;          // 1024
    float* ffn    = w + 5120;          // 4096
    float* S_t    = w + 9216;          // 16*1504 = 24064
    float* Wsum_t = w + 33280;         // 1024*16 = 16384
    float* logits = w + 49664;         // 8192
    float* knew   = w + 57856;         // L*1024
    float* vnew   = w + 66048;         // L*1024
    unsigned short* Ut     = (unsigned short*)(w + 74240);    // 16*1024 bf16  (8192 f)
    unsigned short* Pt     = (unsigned short*)(w + 82432);    // 16*1504 bf16  (12032 f)
    unsigned short* enc_bf = (unsigned short*)(w + 94464);    // 1504*1024 bf16 (770048 f)
    unsigned short* enc_t  = (unsigned short*)(w + 864512);   // 1024*1504 bf16 (770048 f)
    float* part   = w + 1634560;       // up to 4*8192

    k_embed<<<4, 256, 0, stream>>>(tok_emb, pos_emb, input_id, stepp, x);
    k_prep_enc<<<dim3(H_ / 32, ENCP_ / 32), 256, 0, stream>>>(enc, enc_bf, enc_t);

    for (int l = 0; l < L_; ++l) {
        const size_t mo = (size_t)l * H_ * H_;
        // ---- self attention ----
        k_layernorm<<<1, 256, 0, stream>>>(x, ln1_g + l * H_, ln1_b + l * H_, h);
        gemv(stream, h, Wq + mo, part, q,             H_, H_, 4, 0);
        gemv(stream, h, Wk + mo, part, knew + l * H_, H_, H_, 4, 0);
        gemv(stream, h, Wv + mo, part, vnew + l * H_, H_, H_, 4, 0);
        k_self_attn<<<NH_, 256, 0, stream>>>(q,
            cache_k + (size_t)l * NH_ * SMAX_ * DH_,
            cache_v + (size_t)l * NH_ * SMAX_ * DH_,
            knew + l * H_, vnew + l * H_, stepp, t1024);
        gemv(stream, t1024, Wo + mo, part, x, H_, H_, 4, 1);
        // ---- cross attention (reassociated: never materialize ek/ev) ----
        k_layernorm<<<1, 256, 0, stream>>>(x, ln2_g + l * H_, ln2_b + l * H_, h);
        gemv(stream, h, Cq + mo, part, qc, H_, H_, 4, 0);
        k_crossU<<<64, 256, 0, stream>>>(qc, Ck + mo, Ut);
        k_scores_wmma<<<ENCP_ / 16, 32, 0, stream>>>(enc_bf, Ut, S_t);
        k_softmax_head<<<NH_, 256, 0, stream>>>(S_t, cmask, Pt);
        k_wsum_wmma<<<H_ / 16, 32, 0, stream>>>(Pt, enc_t, Wsum_t);
        k_crossV<<<4, 256, 0, stream>>>(Wsum_t, Cv + mo, t1024);
        gemv(stream, t1024, Co + mo, part, x, H_, H_, 4, 1);
        // ---- FFN ----
        k_layernorm<<<1, 256, 0, stream>>>(x, ln3_g + l * H_, ln3_b + l * H_, h);
        gemv(stream, h,   W1 + (size_t)l * H_ * FFN_, part, ffn, H_,   FFN_, 4, 2);
        gemv(stream, ffn, W2 + (size_t)l * FFN_ * H_, part, x,   FFN_, H_,   8, 1);
    }

    // ---- final LN + logits + log_softmax ----
    k_layernorm<<<1, 256, 0, stream>>>(x, lnf_g, lnf_b, h);
    gemv(stream, h, Wout, part, logits, H_, VOCAB_, 4, 0);
    k_logsoftmax<<<1, 256, 0, stream>>>(logits, (float*)d_out);

    // ---- cache restack: nk, nv after the 8192 logits ----
    float* outk = (float*)d_out + VOCAB_;
    float* outv = outk + (size_t)L_ * NH_ * SMAX_ * DH_;
    k_cache<<<8192, 256, 0, stream>>>(cache_k, cache_v, knew, vnew, stepp, outk, outv);
}
